// GCN_2_23459111371162
// MI455X (gfx1250) — matrix operand order repaired and checked by reading, not compiled, not run on hardware
//
#include <hip/hip_runtime.h>
#include <hip/hip_bf16.h>

typedef __attribute__((ext_vector_type(16))) _Float16 v16h;
typedef __attribute__((ext_vector_type(8)))  float    v8f;

#define IN_F  128
#define OUT_F 64

__device__ __forceinline__ void atomAddF32(float* p, float v) {
  // lowers to global_atomic_add_f32 (no CAS loop)
  unsafeAtomicAdd(p, v);
}

// ---------------------------------------------------------------- zero scratch
__global__ __launch_bounds__(256) void k_zero(float* agg, unsigned* cnt,
                                              int n_agg, int n_cnt) {
  int i = blockIdx.x * 256 + threadIdx.x;
  if (i < n_agg) agg[i] = 0.0f;
  if (i < n_cnt) cnt[i] = 0u;
}

// ---------------------------------------------------------------- degree histogram
__global__ __launch_bounds__(256) void k_degrees(const int* __restrict__ src,
                                                 const int* __restrict__ dst,
                                                 unsigned* __restrict__ cnt_out,
                                                 unsigned* __restrict__ cnt_in,
                                                 int n_edges) {
  int e = blockIdx.x * 256 + threadIdx.x;
  if (e < n_edges) {
    atomicAdd(&cnt_out[src[e]], 1u);
    atomicAdd(&cnt_in[dst[e]], 1u);
  }
}

// ---------------------------------------------------------------- counts -> rsqrt(max(c,1))
__global__ __launch_bounds__(256) void k_rsqrt(const unsigned* __restrict__ cnt,
                                               float* __restrict__ rs, int n) {
  int i = blockIdx.x * 256 + threadIdx.x;
  if (i < n) {
    unsigned c = cnt[i];
    rs[i] = rsqrtf((float)(c < 1u ? 1u : c));
  }
}

// ---------------------------------------------------------------- pack W into per-lane WMMA B fragments (f16)
// packed[((c*4 + t)*32 + lane)*16 + i] = f16( W[(c*32 + (lane>=16)*8 + (i&7) + (i>>3)*16)*64 + t*16 + (lane&15)] )
// c = K chunk (0..3), t = 16-col tile (0..3). 8192 halfs = 16 KB.
__global__ __launch_bounds__(256) void k_pack_w(const float* __restrict__ Wm,
                                                _Float16* __restrict__ packed) {
  int tid = blockIdx.x * 256 + threadIdx.x;   // 0..8191
  if (tid < 4 * 4 * 32 * 16) {
    const int i    = tid & 15;
    const int lane = (tid >> 4) & 31;
    const int t    = (tid >> 9) & 3;
    const int c    = tid >> 11;
    const int row  = c * 32 + (lane >> 4) * 8 + (i & 7) + (i >> 3) * 16;   // K index
    const int col  = t * 16 + (lane & 15);            // N index
    packed[tid] = (_Float16)Wm[row * OUT_F + col];
  }
}

// ---------------------------------------------------------------- XW = (X @ W) * rsqrt(deg_out)[row]
// One wave -> 16 rows x 64 cols. 4 accumulator tiles, K in chunks of 32.
__global__ __launch_bounds__(256) void k_gemm_wmma(const float* __restrict__ X,
                                                   const _Float16* __restrict__ Wpk,
                                                   const float* __restrict__ rs_out,
                                                   float* __restrict__ XW,
                                                   int n_nodes) {
  const int wave = (blockIdx.x * 256 + threadIdx.x) >> 5;
  const int lane = threadIdx.x & 31;
  const int row_base = wave << 4;
  if (row_base >= n_nodes) return;           // wave-uniform: EXEC stays all-1s for WMMA

  const int m  = lane & 15;
  const int hi = lane >> 4;                  // which half-wave
  int row = row_base + m;
  if (row >= n_nodes) row = n_nodes - 1;     // tail clamp (n_nodes % 16 == 0 in practice)
  const float4* __restrict__ x4 = (const float4*)(X + (size_t)row * IN_F);

  v8f acc[4] = {v8f{}, v8f{}, v8f{}, v8f{}};

  #pragma unroll
  for (int c = 0; c < 4; ++c) {              // K chunk: kk = 32*c
    // A fragment, 16-bit 16x32 layout:
    //   lane<16 : K = kk+{0..7, 16..23};  lane>=16 : K = kk+{8..15, 24..31}
    const int q = c * 8 + hi * 2;            // float4 index of kk + hi*8
    const float4 xa = x4[q];
    const float4 xb = x4[q + 1];
    const float4 xc = x4[q + 4];
    const float4 xd = x4[q + 5];
    v16h a;
    a[0]  = (_Float16)xa.x;  a[1]  = (_Float16)xa.y;
    a[2]  = (_Float16)xa.z;  a[3]  = (_Float16)xa.w;
    a[4]  = (_Float16)xb.x;  a[5]  = (_Float16)xb.y;
    a[6]  = (_Float16)xb.z;  a[7]  = (_Float16)xb.w;
    a[8]  = (_Float16)xc.x;  a[9]  = (_Float16)xc.y;
    a[10] = (_Float16)xc.z;  a[11] = (_Float16)xc.w;
    a[12] = (_Float16)xd.x;  a[13] = (_Float16)xd.y;
    a[14] = (_Float16)xd.z;  a[15] = (_Float16)xd.w;

    #pragma unroll
    for (int t = 0; t < 4; ++t) {
      const v16h bfr = *(const v16h*)(Wpk + (((size_t)(c * 4 + t) * 32 + lane) << 4));
      acc[t] = __builtin_amdgcn_wmma_f32_16x16x32_f16(
          /*neg_a=*/false, a, /*neg_b=*/false, bfr,
          /*c_mod=*/(short)0, acc[t], /*reuse_a=*/false, /*reuse_b=*/false);
    }
  }

  // C/D layout: VGPR r -> row (row_base + r + 8*hi), col = t*16 + m. Scale by rsqrt(deg_out).
  #pragma unroll
  for (int r = 0; r < 8; ++r) {
    const int orow = row_base + r + hi * 8;
    if (orow < n_nodes) {
      const float s = rs_out[orow];
      float* op = XW + (size_t)orow * OUT_F + m;
      op[0]  = acc[0][r] * s;
      op[16] = acc[1][r] * s;
      op[32] = acc[2][r] * s;
      op[48] = acc[3][r] * s;
    }
  }
}

// ---------------------------------------------------------------- agg[dst] += XW_scaled[src]
// One wave per edge; lane handles 2 of the 64 output features -> coalesced 256B gather.
__global__ __launch_bounds__(256) void k_scatter(const int* __restrict__ src,
                                                 const int* __restrict__ dst,
                                                 const float* __restrict__ XW,
                                                 float* __restrict__ agg,
                                                 int n_edges) {
  const int tid  = blockIdx.x * 256 + threadIdx.x;
  const int e    = tid >> 5;
  const int lane = tid & 31;
  if (e >= n_edges) return;
  const int s = src[e];
  const int d = dst[e];
  const float2 v = *(const float2*)(XW + ((size_t)s << 6) + lane * 2);
  float* ap = agg + ((size_t)d << 6) + lane * 2;
  atomAddF32(ap,     v.x);
  atomAddF32(ap + 1, v.y);
}

// ---------------------------------------------------------------- out = relu(agg * rsqrt(deg_in) + b)
__global__ __launch_bounds__(256) void k_finalize(const float* __restrict__ agg,
                                                  const float* __restrict__ rs_in,
                                                  const float* __restrict__ bias,
                                                  float* __restrict__ out, int total) {
  int i = blockIdx.x * 256 + threadIdx.x;
  if (i < total) {
    float h = agg[i] * rs_in[i >> 6] + bias[i & 63];
    out[i] = h > 0.0f ? h : 0.0f;
  }
}

extern "C" void kernel_launch(void* const* d_in, const int* in_sizes, int n_in,
                              void* d_out, int out_size, void* d_ws, size_t ws_size,
                              hipStream_t stream) {
  const float* in_feat = (const float*)d_in[0];
  const int*   src     = (const int*)d_in[1];
  const int*   dst     = (const int*)d_in[2];
  const float* W       = (const float*)d_in[3];
  const float* bias    = (const float*)d_in[4];
  float*       out     = (float*)d_out;

  const int n_nodes = in_sizes[0] / IN_F;
  const int n_edges = in_sizes[1];
  const int n_feat_total = n_nodes * OUT_F;

  // workspace layout (all regions stay 32B-aligned for n_nodes=100000)
  float*    agg     = (float*)d_ws;                            // n_nodes*64 f32
  float*    xw      = agg + (size_t)n_feat_total;              // n_nodes*64 f32
  unsigned* cnt     = (unsigned*)(xw + (size_t)n_feat_total);  // 2*n_nodes u32 (out, in)
  float*    rs      = (float*)(cnt + 2 * (size_t)n_nodes);     // 2*n_nodes f32 (out, in)
  _Float16* wpk     = (_Float16*)(rs + 2 * (size_t)n_nodes);   // 8192 f16 packed B frags
  unsigned* cnt_out = cnt;
  unsigned* cnt_in  = cnt + n_nodes;
  float*    rs_out  = rs;
  float*    rs_in   = rs + n_nodes;

  const int zero_n = (n_feat_total > 2 * n_nodes) ? n_feat_total : 2 * n_nodes;

  k_zero<<<(zero_n + 255) / 256, 256, 0, stream>>>(agg, cnt, n_feat_total, 2 * n_nodes);

  k_degrees<<<(n_edges + 255) / 256, 256, 0, stream>>>(src, dst, cnt_out, cnt_in, n_edges);

  k_rsqrt<<<(2 * n_nodes + 255) / 256, 256, 0, stream>>>(cnt, rs, 2 * n_nodes);

  k_pack_w<<<32, 256, 0, stream>>>(W, wpk);

  const int n_waves  = (n_nodes + 15) / 16;         // one wave per 16 rows
  const int gemm_blk = (n_waves + 7) / 8;           // 8 waves per 256-thread block
  k_gemm_wmma<<<gemm_blk, 256, 0, stream>>>(in_feat, wpk, rs_out, xw, n_nodes);

  const long long sc_threads = (long long)n_edges * 32;
  k_scatter<<<(int)((sc_threads + 255) / 256), 256, 0, stream>>>(src, dst, xw, agg, n_edges);

  k_finalize<<<(n_feat_total + 255) / 256, 256, 0, stream>>>(agg, rs_in, bias, out, n_feat_total);
}
